// TriangleAttention_927712936259
// MI455X (gfx1250) — compile-verified, matrix operand order checked by reading, not run on hardware
//
#include <hip/hip_runtime.h>
#include <hip/hip_bf16.h>
#include <math.h>

// Problem dims (fixed by the reference)
#define NN    320
#define DDIM  128
#define HH    4
#define DHEAD 32
#define INNER 128
#define MTOT  (NN * NN)   // 102400 flattened (r, j) rows

typedef __attribute__((ext_vector_type(16))) __bf16 v16bf;
typedef __attribute__((ext_vector_type(8)))  float  v8f;

// ---------- bf16 helpers (raw-bit storage in workspace) ----------
__device__ __forceinline__ unsigned short f2bfbits(float f) {
    union { float f; unsigned u; } in; in.f = f;
    unsigned u = in.u;
    unsigned r = u + 0x7FFFu + ((u >> 16) & 1u);   // round-to-nearest-even
    return (unsigned short)(r >> 16);
}
__device__ __forceinline__ unsigned pack2bf(float lo, float hi) {
    return (unsigned)f2bfbits(lo) | ((unsigned)f2bfbits(hi) << 16);
}
__device__ __forceinline__ float bits2f(unsigned short h) {
    union { unsigned u; float f; } o; o.u = ((unsigned)h) << 16; return o.f;
}

union FragU { unsigned u[8]; v16bf v; };

// 16x32 bf16 fragment from a row-major bf16(bits) source.
// Lane layout per ISA: row = lane%16, K base = (lane/16)*8,
// element pairs at K = {0,2,4,6,16,18,20,22}+base  ->  uint indices {0,1,2,3, 8,9,10,11}.
// Each half is 16 contiguous bytes (merges to b128 when 16B-aligned).
__device__ __forceinline__ v16bf loadfrag(const unsigned short* X, int ld, int row0, int k0, int lane) {
    int row = row0 + (lane & 15);
    int kb  = k0 + ((lane >> 4) << 3);
    const unsigned int* p = (const unsigned int*)(X + (size_t)row * ld + kb);
    FragU f;
    f.u[0] = p[0];  f.u[1] = p[1];  f.u[2] = p[2];  f.u[3] = p[3];
    f.u[4] = p[8];  f.u[5] = p[9];  f.u[6] = p[10]; f.u[7] = p[11];
    return f.v;
}
__device__ __forceinline__ v8f wmma_bf16(v16bf a, v16bf b, v8f c) {
    return __builtin_amdgcn_wmma_f32_16x16x32_bf16(false, a, false, b, (short)0, c, false, false);
}

// ---------------- Kernel 0a: bulk convert pairwise x (f32) -> bf16 workspace ----------------
__global__ __launch_bounds__(256) void ta_cvt_x(const float* __restrict__ x,
                                                unsigned int* __restrict__ xb) {
    size_t t = (size_t)blockIdx.x * 256 + threadIdx.x;   // each thread: 8 floats
    const float4* p = (const float4*)x + t * 2;
    float4 a = p[0], b = p[1];
    uint4 o;
    o.x = pack2bf(a.x, a.y); o.y = pack2bf(a.z, a.w);
    o.z = pack2bf(b.x, b.y); o.w = pack2bf(b.z, b.w);
    ((uint4*)xb)[t] = o;
}

// ---------------- Kernel 0b: weights -> transposed bf16 [n][k]; Wb -> padded 16x128 ----------------
__global__ __launch_bounds__(128) void ta_prep_w(
    const float* __restrict__ Wq, const float* __restrict__ Wk,
    const float* __restrict__ Wv, const float* __restrict__ Wg,
    const float* __restrict__ Wo, const float* __restrict__ Wb,
    unsigned short* __restrict__ Wt,   // 5 x [128][128], order q,k,v,g,o
    unsigned short* __restrict__ Wbt)  // [16][128], rows >= 4 zero
{
    int blk = blockIdx.x;
    int k   = threadIdx.x;             // 0..127
    if (blk < 640) {
        int m = blk >> 7;              // matrix 0..4
        int n = blk & 127;
        const float* W = (m == 0) ? Wq : (m == 1) ? Wk : (m == 2) ? Wv : (m == 3) ? Wg : Wo;
        Wt[(size_t)m * 16384 + n * 128 + k] = f2bfbits(W[k * 128 + n]);
    } else {
        int n = blk - 640;             // 0..15
        Wbt[n * 128 + k] = (n < HH) ? f2bfbits(Wb[k * HH + n]) : (unsigned short)0;
    }
}

// ---------------- Kernel 1: bias[h,i,j] = x . Wb  via WMMA (N padded to 16) ----------------
__global__ __launch_bounds__(32) void ta_bias(const unsigned short* __restrict__ Xbf,
                                              const unsigned short* __restrict__ Wbt,
                                              float* __restrict__ bias) {
    int mt   = blockIdx.x;
    int lane = threadIdx.x & 31;
    int m0 = mt * 16;
    v8f c = {};
#pragma unroll
    for (int kc = 0; kc < 4; ++kc) {
        v16bf a = loadfrag(Xbf, DDIM, m0, kc * 32, lane);
        v16bf b = loadfrag(Wbt, DDIM, 0, kc * 32, lane);   // B^T rows = heads
        c = wmma_bf16(a, b, c);
    }
#pragma unroll
    for (int p = 0; p < 8; ++p) {
        int row = m0 + p + ((lane >> 4) << 3);
        int col = lane & 15;
        if (col < HH) bias[(size_t)col * MTOT + row] = c[p];
    }
}

// ---------------- Kernel 2: Q/K/V/G projections (V stored transposed [r,e,j]) ----------------
__global__ __launch_bounds__(256) void ta_proj(
    const unsigned short* __restrict__ Xbf, const unsigned short* __restrict__ Wt,
    unsigned short* __restrict__ Q, unsigned short* __restrict__ K,
    unsigned short* __restrict__ Vt, unsigned short* __restrict__ G) {
    int mt   = blockIdx.x;           // 16-row M-tile
    int tid  = threadIdx.x;
    int lane = tid & 31;
    int wave = tid >> 5;             // 0..7 -> one 16-col N-tile
    int m0 = mt * 16;
    int n0 = wave * 16;
    int r  = m0 / NN;                // all 16 rows share r (320 % 16 == 0)
    int j0 = m0 % NN;

    v16bf a[4];
#pragma unroll
    for (int kc = 0; kc < 4; ++kc)
        a[kc] = loadfrag(Xbf, DDIM, m0, kc * 32, lane);

#pragma unroll
    for (int mat = 0; mat < 4; ++mat) {           // q,k,v,g
        const unsigned short* Wm = Wt + (size_t)mat * 16384;
        v8f c = {};
#pragma unroll
        for (int kc = 0; kc < 4; ++kc)
            c = wmma_bf16(a[kc], loadfrag(Wm, DDIM, n0, kc * 32, lane), c);
#pragma unroll
        for (int p = 0; p < 8; ++p) {
            int rl  = p + ((lane >> 4) << 3);
            int col = n0 + (lane & 15);
            if (mat == 2) {
                // V transposed: Vt[(r*INNER + e)*NN + j]
                Vt[((size_t)r * INNER + col) * NN + (j0 + rl)] = f2bfbits(c[p]);
            } else {
                float val = c[p];
                if (mat == 3) val = 1.0f / (1.0f + __expf(-val));   // sigmoid gate
                unsigned short* D = (mat == 0) ? Q : (mat == 1) ? K : G;
                D[(size_t)(m0 + rl) * INNER + col] = f2bfbits(val);
            }
        }
    }
}

// ---------------- Kernel 3: per-(i-tile, row, head) attention ----------------
__global__ __launch_bounds__(64) void ta_attn(
    const unsigned short* __restrict__ Q, const unsigned short* __restrict__ K,
    const unsigned short* __restrict__ Vt, const unsigned short* __restrict__ G,
    const float* __restrict__ bias, const bool* __restrict__ mask,
    unsigned short* __restrict__ O) {
    __shared__ __attribute__((aligned(16))) float          simS[16 * NN];   // 20 KB
    __shared__ __attribute__((aligned(16))) unsigned short attnS[16 * NN];  // 10 KB
    __shared__ float red[64];

    int it   = blockIdx.x;    // i-tile 0..19
    int r    = blockIdx.y;    // row   0..319
    int h    = blockIdx.z;    // head  0..3
    int tid  = threadIdx.x;
    int lane = tid & 31;
    int wave = tid >> 5;      // 0..1

    const float scale = 0.17677669529663687f;    // DH^-0.5

    // -------- Phase 1: sim[16 x 320] = (Q Kt)*scale + bias, masked --------
    v16bf qf = loadfrag(Q, INNER, r * NN + it * 16, h * DHEAD, lane);
    for (int t = 0; t < 10; ++t) {
        int jt = wave * 10 + t;
        v16bf kf = loadfrag(K, INNER, r * NN + jt * 16, h * DHEAD, lane);
        v8f c = {};
        c = wmma_bf16(qf, kf, c);
#pragma unroll
        for (int p = 0; p < 8; ++p) {
            int il = p + ((lane >> 4) << 3);
            int jg = jt * 16 + (lane & 15);
            float v = c[p] * scale
                    + bias[((size_t)h * NN + (it * 16 + il)) * NN + jg];
            if (!mask[jg]) v = -1e9f;
            simS[il * NN + jg] = v;
        }
    }
    __syncthreads();

    // -------- Phase 2: softmax over j, 4 threads per row --------
    {
        int srow = tid >> 2, sq = tid & 3;
        float* rp = simS + srow * NN;
        int jb = sq * 80;
        float mx = -3.4e38f;
        for (int j = jb; j < jb + 80; ++j) mx = fmaxf(mx, rp[j]);
        red[tid] = mx;
        __syncthreads();
        mx = fmaxf(fmaxf(red[srow * 4 + 0], red[srow * 4 + 1]),
                   fmaxf(red[srow * 4 + 2], red[srow * 4 + 3]));
        float s = 0.f;
        for (int j = jb; j < jb + 80; ++j) { float e = __expf(rp[j] - mx); rp[j] = e; s += e; }
        __syncthreads();
        red[tid] = s;
        __syncthreads();
        s = red[srow * 4 + 0] + red[srow * 4 + 1] + red[srow * 4 + 2] + red[srow * 4 + 3];
        float inv = 1.0f / s;
        unsigned int* ap = (unsigned int*)(attnS + srow * NN + jb);
        for (int u = 0; u < 40; ++u)
            ap[u] = pack2bf(rp[jb + 2 * u] * inv, rp[jb + 2 * u + 1] * inv);
    }
    __syncthreads();

    // -------- Phase 3: o = attn @ V (wave owns one 16-wide e-tile), gate, store --------
    int et = wave;
    const unsigned short* Vb = Vt + ((size_t)r * INNER + h * DHEAD + et * 16) * NN;
    v8f c = {};
    for (int kc = 0; kc < 10; ++kc) {                 // j chunks of 32
        v16bf af = loadfrag(attnS, NN, 0, kc * 32, lane);  // A from LDS
        v16bf bf = loadfrag(Vb,   NN, 0, kc * 32, lane);   // B^T rows = e, contiguous j
        c = wmma_bf16(af, bf, c);
    }
#pragma unroll
    for (int p = 0; p < 8; ++p) {
        int il  = p + ((lane >> 4) << 3);
        int ig  = it * 16 + il;
        int col = h * DHEAD + et * 16 + (lane & 15);
        float g = bits2f(G[(size_t)(r * NN + ig) * INNER + col]);   // sigmoid done
        O[(size_t)(r * NN + ig) * INNER + col] = f2bfbits(c[p] * g);
    }
}

// ---------------- Kernel 4: out = (o*gate) @ Wo, f32 output ----------------
__global__ __launch_bounds__(256) void ta_out(const unsigned short* __restrict__ O,
                                              const unsigned short* __restrict__ Wto,
                                              float* __restrict__ out) {
    int mt   = blockIdx.x;
    int tid  = threadIdx.x;
    int lane = tid & 31;
    int wave = tid >> 5;          // 0..7
    int m0 = mt * 16;
    int n0 = wave * 16;

    v16bf a[4];
#pragma unroll
    for (int kc = 0; kc < 4; ++kc)
        a[kc] = loadfrag(O, INNER, m0, kc * 32, lane);

    v8f c = {};
#pragma unroll
    for (int kc = 0; kc < 4; ++kc)
        c = wmma_bf16(a[kc], loadfrag(Wto, DDIM, n0, kc * 32, lane), c);

#pragma unroll
    for (int p = 0; p < 8; ++p) {
        int row = m0 + p + ((lane >> 4) << 3);
        int col = n0 + (lane & 15);
        out[(size_t)row * DDIM + col] = c[p];
    }
}

// ---------------- Host side ----------------
extern "C" void kernel_launch(void* const* d_in, const int* in_sizes, int n_in,
                              void* d_out, int out_size, void* d_ws, size_t ws_size,
                              hipStream_t stream) {
    const float* x    = (const float*)d_in[0];
    const bool*  mask = (const bool*) d_in[1];
    const float* Wq   = (const float*)d_in[2];
    const float* Wk   = (const float*)d_in[3];
    const float* Wv   = (const float*)d_in[4];
    const float* Wg   = (const float*)d_in[5];
    const float* Wo   = (const float*)d_in[6];
    const float* Wb   = (const float*)d_in[7];
    float* out = (float*)d_out;

    // Workspace: Xbf, Q, K, G, O, Vt (bf16 [MTOT x 128] each) + bias f32 + weight tables
    const size_t MAT = (size_t)MTOT * INNER;
    unsigned short* Xbf = (unsigned short*)d_ws;
    unsigned short* Qw  = Xbf + MAT;
    unsigned short* Kw  = Qw + MAT;
    unsigned short* Gw  = Kw + MAT;
    unsigned short* Ow  = Gw + MAT;
    unsigned short* Vtw = Ow + MAT;
    float*          biasW = (float*)(Vtw + MAT);
    unsigned short* Wt  = (unsigned short*)(biasW + (size_t)HH * MTOT);  // 5 x 128 x 128
    unsigned short* Wbt = Wt + 5 * 16384;                                // 16 x 128

    ta_cvt_x<<<(MTOT * DDIM / 8 + 255) / 256, 256, 0, stream>>>(x, (unsigned int*)Xbf);
    ta_prep_w<<<656, 128, 0, stream>>>(Wq, Wk, Wv, Wg, Wo, Wb, Wt, Wbt);
    ta_bias<<<MTOT / 16, 32, 0, stream>>>(Xbf, Wbt, biasW);
    ta_proj<<<MTOT / 16, 256, 0, stream>>>(Xbf, Wt, Qw, Kw, Vtw, Gw);
    ta_attn<<<dim3(NN / 16, NN, HH), 64, 0, stream>>>(Qw, Kw, Vtw, Gw, biasW, mask, Ow);
    ta_out<<<MTOT / 16, 256, 0, stream>>>(Ow, Wt + 4 * 16384, out);
}
